// TrainModel_86560770883785
// MI455X (gfx1250) — matrix operand order repaired and checked by reading, not compiled, not run on hardware
//
#include <hip/hip_runtime.h>
#include <hip/hip_bf16.h>

// ---------------------------------------------------------------------------
// Batched GCN (B=2048 graphs, N=116 nodes) for MI455X / gfx1250.
// All matmuls run through v_wmma_f32_16x16x32_f16 (f16 inputs, f32 accum).
// Dims padded 116 -> 128. Intermediates live in d_ws as f16, oriented for
// their consumer: row-major [M][K] when next used as A, transposed [N][K]
// when next used as B (so every fragment load is contiguous 16B runs).
// ---------------------------------------------------------------------------

typedef __attribute__((ext_vector_type(16))) _Float16 v16h;
typedef __attribute__((ext_vector_type(8)))  float    v8f;

#define NB 2048
#define NVAL 116
#define NPAD 128

// -------------------------- adjacency normalization -------------------------
// a_n = D^{-1/2} (A+I) D^{-1/2}, written as zero-padded f16 [128][128].
__global__ void norm_adj_kernel(const float* __restrict__ adj,
                                _Float16* __restrict__ an) {
  int g = blockIdx.x;
  const float* a = adj + (size_t)g * (NVAL * NVAL);
  __shared__ float dinv[NVAL];
  if (threadIdx.x < NVAL) {
    int i = threadIdx.x;
    float s = 1.0f;  // self-loop
    for (int j = 0; j < NVAL; ++j) s += a[i * NVAL + j];
    dinv[i] = rsqrtf(fmaxf(s, 1e-12f));
  }
  __syncthreads();
  _Float16* dst = an + (size_t)g * (NPAD * NPAD);
  for (int idx = threadIdx.x; idx < NPAD * NPAD; idx += blockDim.x) {
    int i = idx >> 7, j = idx & (NPAD - 1);
    float v = 0.f;
    if (i < NVAL && j < NVAL)
      v = (a[i * NVAL + j] + (i == j ? 1.f : 0.f)) * dinv[i] * dinv[j];
    dst[idx] = (_Float16)v;
  }
}

// ------------------------- x -> padded f16 [128][128] -----------------------
__global__ void convert_x_kernel(const float* __restrict__ x,
                                 _Float16* __restrict__ xh) {
  int g = blockIdx.x;
  const float* s = x + (size_t)g * (NVAL * NVAL);
  _Float16* d = xh + (size_t)g * (NPAD * NPAD);
  for (int idx = threadIdx.x; idx < NPAD * NPAD; idx += blockDim.x) {
    int i = idx >> 7, j = idx & (NPAD - 1);
    d[idx] = (i < NVAL && j < NVAL) ? (_Float16)s[i * NVAL + j] : (_Float16)0.f;
  }
}

// --------- weight f32 [Kv][Nv] -> f16 transposed padded [Npad][Kpad] --------
__global__ void convert_wt_kernel(const float* __restrict__ W,
                                  _Float16* __restrict__ Wt,
                                  int Kv, int Nv, int Kpad, int Npad) {
  int total = Npad * Kpad;
  for (int idx = blockIdx.x * blockDim.x + threadIdx.x; idx < total;
       idx += gridDim.x * blockDim.x) {
    int n = idx / Kpad, k = idx % Kpad;
    Wt[idx] = (k < Kv && n < Nv) ? (_Float16)W[k * Nv + n] : (_Float16)0.f;
  }
}

// ------------------------------ WMMA GEMM -----------------------------------
// D = act(A[M,K] * B[K,N] + bias).  A row-major f16 (lda=Kpad), B supplied
// TRANSPOSED row-major [Npad][Kpad] (ldb=Kpad).  Batched via strides.
// One block = 8 waves; wave w owns N-tile w; grid = (batch, Mpad/16).
// d16mode: 0=none, 1=row-major [M][N] (ld16), 2=transposed [N][M] (ld16).
// Padded rows/cols (>= m_valid / n_valid) are stored as exact zeros so they
// can safely feed later K-dimensions and the mean pool.
__global__ void wmma_gemm_kernel(
    const _Float16* __restrict__ A, int lda, unsigned long long strideA,
    const _Float16* __restrict__ Bt, int ldb, unsigned long long strideB,
    const float* __restrict__ bias, int Kpad, int Npad, int relu,
    int m_valid, int n_valid,
    _Float16* __restrict__ D16, int d16mode, int ld16, unsigned long long stride16,
    float* __restrict__ D32, int ld32, unsigned long long stride32) {
  int g = blockIdx.x;
  int mt = blockIdx.y;
  int wave = threadIdx.x >> 5;
  int lane = threadIdx.x & 31;
  int n0 = wave * 16;
  if (n0 >= Npad) return;  // wave-uniform: EXEC stays all-ones for WMMA

  const _Float16* Ag = A + (size_t)g * strideA;
  const _Float16* Bg = Bt + (size_t)g * strideB;
  int m0 = mt * 16;

  v8f c = {0.f, 0.f, 0.f, 0.f, 0.f, 0.f, 0.f, 0.f};

  for (int k0 = 0; k0 < Kpad; k0 += 32) {
    v16h af, bf;
    {  // A fragment: lane = row m; halves 0..7 -> K=kb..kb+7, 8..15 -> kb+16..
      int m = m0 + (lane & 15);
      int kb = k0 + ((lane & 16) ? 8 : 0);
      const _Float16* p = Ag + (size_t)m * lda + kb;
      if (k0 + 32 < Kpad) __builtin_prefetch((const void*)(p + 32), 0, 1);
#pragma unroll
      for (int e = 0; e < 8; ++e) af[e] = p[e];
#pragma unroll
      for (int e = 0; e < 8; ++e) af[8 + e] = p[16 + e];
    }
    {  // B fragment from transposed storage: lane = col n; halves 0..7 -> K=kb..kb+7, 8..15 -> kb+16..
      int n = n0 + (lane & 15);
      int kb = k0 + ((lane & 16) ? 8 : 0);
      const _Float16* p = Bg + (size_t)n * ldb + kb;
#pragma unroll
      for (int e = 0; e < 8; ++e) bf[e] = p[e];
#pragma unroll
      for (int e = 0; e < 8; ++e) bf[8 + e] = p[16 + e];
    }
    c = __builtin_amdgcn_wmma_f32_16x16x32_f16(false, af, false, bf,
                                               (short)0, c, false, false);
  }

  // C/D layout: VGPR r holds (M=m0+r, N) for lanes 0-15, (M=m0+8+r, N) 16-31.
  int n = n0 + (lane & 15);
  int moff = (lane & 16) ? 8 : 0;
  float bv = (bias != nullptr && n < n_valid) ? bias[n] : 0.f;
#pragma unroll
  for (int r = 0; r < 8; ++r) {
    int m = m0 + moff + r;
    float v = c[r] + bv;
    if (relu) v = fmaxf(v, 0.f);
    bool ok = (m < m_valid) && (n < n_valid);
    _Float16 h = ok ? (_Float16)v : (_Float16)0.f;
    if (d16mode == 1)
      D16[(size_t)g * stride16 + (size_t)m * ld16 + n] = h;
    else if (d16mode == 2)
      D16[(size_t)g * stride16 + (size_t)n * ld16 + m] = h;
    if (D32 != nullptr && ok)
      D32[(size_t)g * stride32 + (size_t)m * ld32 + n] = v;
  }
}

// --------------------- mean pool over 116 nodes -> f16 ----------------------
__global__ void pool_mean_kernel(const _Float16* __restrict__ H,  // [B][128][64]
                                 _Float16* __restrict__ pooled) { // [B][64]
  int g = blockIdx.x;
  int f = threadIdx.x;  // 64 threads
  const _Float16* h = H + (size_t)g * (NPAD * 64);
  float s = 0.f;
  for (int i = 0; i < NVAL; ++i) s += (float)h[i * 64 + f];
  pooled[(size_t)g * 64 + f] = (_Float16)(s * (1.f / (float)NVAL));
}

// ------------------- final 64->2 linear + softmax ---------------------------
__global__ void head_softmax_kernel(const _Float16* __restrict__ L2,  // [B][64]
                                    const float* __restrict__ m3w,    // [64][2]
                                    const float* __restrict__ m3b,
                                    float* __restrict__ out) {        // [B][2]
  int g = blockIdx.x * blockDim.x + threadIdx.x;
  if (g >= NB) return;
  const _Float16* l = L2 + (size_t)g * 64;
  float l0 = m3b[0], l1 = m3b[1];
  for (int k = 0; k < 64; ++k) {
    float v = (float)l[k];
    l0 += v * m3w[k * 2 + 0];
    l1 += v * m3w[k * 2 + 1];
  }
  float mx = fmaxf(l0, l1);
  float e0 = __expf(l0 - mx), e1 = __expf(l1 - mx);
  float s = e0 + e1;
  out[(size_t)g * 2 + 0] = e0 / s;
  out[(size_t)g * 2 + 1] = e1 / s;
}

// ---------------------------------------------------------------------------
static inline void launch_gemm(hipStream_t stream, int batch, int Mpad,
                               const _Float16* A, int lda, size_t sA,
                               const _Float16* Bt, int ldb, size_t sB,
                               const float* bias, int Kpad, int Npad, int relu,
                               int mv, int nv,
                               _Float16* D16, int d16mode, int ld16, size_t s16,
                               float* D32, int ld32, size_t s32) {
  dim3 grid(batch, Mpad / 16);
  wmma_gemm_kernel<<<grid, 256, 0, stream>>>(
      A, lda, (unsigned long long)sA, Bt, ldb, (unsigned long long)sB, bias,
      Kpad, Npad, relu, mv, nv, D16, d16mode, ld16, (unsigned long long)s16,
      D32, ld32, (unsigned long long)s32);
}

extern "C" void kernel_launch(void* const* d_in, const int* in_sizes, int n_in,
                              void* d_out, int out_size, void* d_ws, size_t ws_size,
                              hipStream_t stream) {
  (void)in_sizes; (void)n_in; (void)out_size; (void)ws_size;
  const float* x     = (const float*)d_in[0];
  const float* adj   = (const float*)d_in[1];
  const float* adja  = (const float*)d_in[2];
  const float* w1    = (const float*)d_in[3];
  const float* b1    = (const float*)d_in[4];
  const float* w2    = (const float*)d_in[5];
  const float* b2    = (const float*)d_in[6];
  const float* fc1w  = (const float*)d_in[7];
  const float* fc1b  = (const float*)d_in[8];
  const float* fc2w  = (const float*)d_in[9];
  const float* fc2b  = (const float*)d_in[10];
  const float* g1w   = (const float*)d_in[11];
  const float* g1b   = (const float*)d_in[12];
  const float* g2w   = (const float*)d_in[13];
  const float* g2b   = (const float*)d_in[14];
  const float* m1w   = (const float*)d_in[15];
  const float* m1b   = (const float*)d_in[16];
  const float* m2w   = (const float*)d_in[17];
  const float* m2b   = (const float*)d_in[18];
  const float* m3w   = (const float*)d_in[19];
  const float* m3b   = (const float*)d_in[20];

  float* out = (float*)d_out;
  float* o_z1 = out;                                 // [2048][128]
  float* o_z2 = o_z1 + (size_t)NB * 128;             // [2048][128]
  float* o_sm = o_z2 + (size_t)NB * 128;             // [2048][2]
  float* o_x  = o_sm + (size_t)NB * 2;               // [2048*116][116]
  float* o_nx = o_x + (size_t)NB * NVAL * NVAL;      // [2048*116][116]

  // -------- workspace carve-up (f16 halves), ~405 MB total ----------
  _Float16* wsb = (_Float16*)d_ws;
  size_t off = 0;
  auto take = [&](size_t n) {
    _Float16* p = wsb + off;
    off += (n + 255) & ~(size_t)255;
    return p;
  };
  const size_t SGG = (size_t)NPAD * NPAD;  // 16384 per graph
  const size_t SGH = (size_t)NPAD * 64;    // 8192 per graph
  _Float16* anorm = take((size_t)NB * SGG);  // a_n  f16 padded (A operand)
  _Float16* aaug  = take((size_t)NB * SGG);  // a_aug_n
  _Float16* xh    = take((size_t)NB * SGG);  // x    f16 padded (A operand)
  _Float16* bufA  = take((size_t)NB * SGG);  // transposed [N][K] scratch (B operand)
  _Float16* bufB  = take((size_t)NB * SGG);  // row-major scratch (A operand)
  _Float16* bufC  = take((size_t)NB * SGH);  // transposed [64][128] scratch
  _Float16* bufD  = take((size_t)NB * SGH);  // row-major [128][64] scratch
  _Float16* pool1 = take((size_t)NB * 64);
  _Float16* pool2 = take((size_t)NB * 64);
  _Float16* bufP  = take((size_t)NB * 128);  // head intermediate [2048][128]
  _Float16* bufL  = take((size_t)NB * 64);   // head L2 [2048][64]
  _Float16* w1t  = take(128 * 128);
  _Float16* w2t  = take(64 * 128);
  _Float16* g1t  = take(64 * 128);
  _Float16* g2t  = take(128 * 64);
  _Float16* fc1t = take(128 * 64);
  _Float16* fc2t = take(128 * 128);
  _Float16* m1t  = take(128 * 64);
  _Float16* m2t  = take(64 * 128);

  // -------- raw x copy straight into output slot 4 ----------
  hipMemcpyAsync(o_x, x, (size_t)NB * NVAL * NVAL * sizeof(float),
                 hipMemcpyDeviceToDevice, stream);

  // -------- precompute: normalized adjacencies, f16 x, f16 weights ----------
  norm_adj_kernel<<<NB, 256, 0, stream>>>(adj, anorm);
  norm_adj_kernel<<<NB, 256, 0, stream>>>(adja, aaug);
  convert_x_kernel<<<NB, 256, 0, stream>>>(x, xh);
  convert_wt_kernel<<<64, 256, 0, stream>>>(w1,  w1t,  116, 128, 128, 128);
  convert_wt_kernel<<<32, 256, 0, stream>>>(w2,  w2t,  128,  64, 128,  64);
  convert_wt_kernel<<<32, 256, 0, stream>>>(g1w, g1t,  116,  64, 128,  64);
  convert_wt_kernel<<<32, 256, 0, stream>>>(g2w, g2t,   64, 116,  64, 128);
  convert_wt_kernel<<<32, 256, 0, stream>>>(fc1w, fc1t, 64, 128,  64, 128);
  convert_wt_kernel<<<64, 256, 0, stream>>>(fc2w, fc2t, 128, 128, 128, 128);
  convert_wt_kernel<<<32, 256, 0, stream>>>(m1w, m1t,   64, 128,  64, 128);
  convert_wt_kernel<<<32, 256, 0, stream>>>(m2w, m2t,  128,  64, 128,  64);

  // -------- feature extractor on (x, a_n) ----------
  // XW1^T = (x @ w1)^T        -> bufA [128n][128m]
  launch_gemm(stream, NB, NPAD, xh, 128, SGG, w1t, 128, 0, nullptr, 128, 128, 0,
              116, 128, bufA, 2, 128, SGG, nullptr, 0, 0);
  // H1 = relu(a_n @ XW1 + b1) -> bufB row-major [128][128]
  launch_gemm(stream, NB, NPAD, anorm, 128, SGG, bufA, 128, SGG, b1, 128, 128, 1,
              116, 128, bufB, 1, 128, SGG, nullptr, 0, 0);
  // H1W2^T = (H1 @ w2)^T      -> bufC [64n][128m]
  launch_gemm(stream, NB, NPAD, bufB, 128, SGG, w2t, 128, 0, nullptr, 128, 64, 0,
              116, 64, bufC, 2, 128, SGH, nullptr, 0, 0);
  // H2 = relu(a_n @ H1W2 + b2) -> bufD row-major [128][64]
  launch_gemm(stream, NB, NPAD, anorm, 128, SGG, bufC, 128, SGH, b2, 128, 64, 1,
              116, 64, bufD, 1, 64, SGH, nullptr, 0, 0);
  pool_mean_kernel<<<NB, 64, 0, stream>>>(bufD, pool1);

  // -------- augmentation: new_x = a_n @ (relu(a_n @ (x@g1) + g1b) @ g2) + g2b
  // XG1^T -> bufA [64n][128m]
  launch_gemm(stream, NB, NPAD, xh, 128, SGG, g1t, 128, 0, nullptr, 128, 64, 0,
              116, 64, bufA, 2, 128, SGG, nullptr, 0, 0);
  // T = relu(a_n @ XG1 + g1b) -> bufB row-major [128][64]
  launch_gemm(stream, NB, NPAD, anorm, 128, SGG, bufA, 128, SGG, g1b, 128, 64, 1,
              116, 64, bufB, 1, 64, SGG, nullptr, 0, 0);
  // TG2^T = (T @ g2)^T        -> bufA [128n][128m]   (K=64)
  launch_gemm(stream, NB, NPAD, bufB, 64, SGG, g2t, 64, 0, nullptr, 64, 128, 0,
              116, 116, bufA, 2, 128, SGG, nullptr, 0, 0);
  // new_x = a_n @ TG2 + g2b   -> bufB row-major f16 + f32 output slot 5
  launch_gemm(stream, NB, NPAD, anorm, 128, SGG, bufA, 128, SGG, g2b, 128, 128, 0,
              116, 116, bufB, 1, 128, SGG, o_nx, NVAL, (size_t)NVAL * NVAL);

  // -------- feature extractor on (new_x, a_aug_n) ----------
  launch_gemm(stream, NB, NPAD, bufB, 128, SGG, w1t, 128, 0, nullptr, 128, 128, 0,
              116, 128, bufA, 2, 128, SGG, nullptr, 0, 0);
  launch_gemm(stream, NB, NPAD, aaug, 128, SGG, bufA, 128, SGG, b1, 128, 128, 1,
              116, 128, bufB, 1, 128, SGG, nullptr, 0, 0);
  launch_gemm(stream, NB, NPAD, bufB, 128, SGG, w2t, 128, 0, nullptr, 128, 64, 0,
              116, 64, bufC, 2, 128, SGH, nullptr, 0, 0);
  launch_gemm(stream, NB, NPAD, aaug, 128, SGG, bufC, 128, SGH, b2, 128, 64, 1,
              116, 64, bufD, 1, 64, SGH, nullptr, 0, 0);
  pool_mean_kernel<<<NB, 64, 0, stream>>>(bufD, pool2);

  // -------- fc heads (batch-as-M GEMMs, M=2048) ----------
  // z1_after = relu(relu(pool1 @ fc1 + b) @ fc2 + b)
  launch_gemm(stream, 1, NB, pool1, 64, 0, fc1t, 64, 0, fc1b, 64, 128, 1,
              NB, 128, bufP, 1, 128, 0, nullptr, 0, 0);
  launch_gemm(stream, 1, NB, bufP, 128, 0, fc2t, 128, 0, fc2b, 128, 128, 1,
              NB, 128, nullptr, 0, 0, 0, o_z1, 128, 0);
  // z2_after
  launch_gemm(stream, 1, NB, pool2, 64, 0, fc1t, 64, 0, fc1b, 64, 128, 1,
              NB, 128, bufP, 1, 128, 0, nullptr, 0, 0);
  launch_gemm(stream, 1, NB, bufP, 128, 0, fc2t, 128, 0, fc2b, 128, 128, 1,
              NB, 128, nullptr, 0, 0, 0, o_z2, 128, 0);
  // classifier: relu(relu(pool1 @ m1 + b) @ m2 + b) @ m3 + b -> softmax
  launch_gemm(stream, 1, NB, pool1, 64, 0, m1t, 64, 0, m1b, 64, 128, 1,
              NB, 128, bufP, 1, 128, 0, nullptr, 0, 0);
  launch_gemm(stream, 1, NB, bufP, 128, 0, m2t, 128, 0, m2b, 128, 64, 1,
              NB, 64, bufL, 1, 64, 0, nullptr, 0, 0);
  head_softmax_kernel<<<(NB + 255) / 256, 256, 0, stream>>>(bufL, m3w, m3b, o_sm);
}
